// LlamaAttention_88759794139636
// MI455X (gfx1250) — matrix operand order repaired and checked by reading, not compiled, not run on hardware
//
#include <hip/hip_runtime.h>

typedef __attribute__((ext_vector_type(16))) __bf16 bf16x16;
typedef __attribute__((ext_vector_type(8)))  __bf16 bf16x8;
typedef __attribute__((ext_vector_type(8)))  float  f32x8;

#define NH   32
#define NKV  8
#define HD   128
#define SEQ  2048
#define BATCH 2
#define HID  4096
#define MROWS (BATCH*SEQ)   // 4096

__device__ __forceinline__ __bf16 to_bf16(float f) {
  unsigned u = __builtin_bit_cast(unsigned, f);
  u += 0x7fffu + ((u >> 16) & 1u);               // round to nearest even
  unsigned short h = (unsigned short)(u >> 16);
  return __builtin_bit_cast(__bf16, h);
}

// ---- CDNA5 async global->LDS DMA (ASYNCcnt-tracked), per-lane 16B copy ----
__device__ __forceinline__ void async_ld_b128(void* lds, const void* gsrc) {
  unsigned l = (unsigned)(size_t)lds;            // low 32 bits = LDS byte address
  asm volatile("global_load_async_to_lds_b128 %0, %1, off"
               :: "v"(l), "v"(gsrc) : "memory");
}
__device__ __forceinline__ void wait_async0() {
  asm volatile("s_wait_asynccnt 0x0" ::: "memory");
}

// ---------------- elementwise fp32 -> bf16 ----------------
__global__ void cvt_bf16_kernel(const float* __restrict__ in, __bf16* __restrict__ out, int n) {
  int i = blockIdx.x * blockDim.x + threadIdx.x;
  if (i < n) out[i] = to_bf16(in[i]);
}

// ---------------- tiled bf16 GEMM: C[M,N] = A[M,K] * W[N,K]^T ----------------
// block = 128x128 tile, 256 threads = 8 waves (4x2), wave tile = 32x64
// double-buffered LDS filled by async LDS-DMA, overlapped with WMMA
__global__ __launch_bounds__(256)
void gemm_bf16_kernel(const __bf16* __restrict__ A, const __bf16* __restrict__ W,
                      float* __restrict__ C, int M, int N, int K) {
  __shared__ __bf16 As[2][128 * 32];
  __shared__ __bf16 Ws[2][128 * 32];
  const int tid = threadIdx.x;
  const int lane = tid & 31, w = tid >> 5;
  const int lane16 = lane & 15, hi = lane >> 4;
  const int wm = w & 3, wn = w >> 2;
  const int bm = blockIdx.y * 128, bn = blockIdx.x * 128;

  f32x8 acc[2][4];
  for (int a = 0; a < 2; a++)
    for (int b = 0; b < 4; b++)
      for (int e = 0; e < 8; e++) acc[a][b][e] = 0.f;

  const int row = tid >> 1, half = tid & 1;
  const __bf16* Ag = A + (size_t)(bm + row) * K + half * 16;
  const __bf16* Wg = W + (size_t)(bn + row) * K + half * 16;

  // prologue: DMA first K-slice into buffer 0
  async_ld_b128(&As[0][row * 32 + half * 16],     Ag);
  async_ld_b128(&As[0][row * 32 + half * 16 + 8], Ag + 8);
  async_ld_b128(&Ws[0][row * 32 + half * 16],     Wg);
  async_ld_b128(&Ws[0][row * 32 + half * 16 + 8], Wg + 8);

  int cur = 0;
  for (int kt = 0; kt < K; kt += 32) {
    wait_async0();        // my DMA into buf `cur` done
    __syncthreads();      // everyone's DMA done + everyone done with buf cur^1
    if (kt + 32 < K) {    // DMA next slice into the other buffer while computing
      const int nxt = cur ^ 1;
      async_ld_b128(&As[nxt][row * 32 + half * 16],     Ag + kt + 32);
      async_ld_b128(&As[nxt][row * 32 + half * 16 + 8], Ag + kt + 40);
      async_ld_b128(&Ws[nxt][row * 32 + half * 16],     Wg + kt + 32);
      async_ld_b128(&Ws[nxt][row * 32 + half * 16 + 8], Wg + kt + 40);
    }

    bf16x16 afrag[2];
#pragma unroll
    for (int mi = 0; mi < 2; mi++) {
      const __bf16* ap = &As[cur][(wm * 32 + mi * 16 + lane16) * 32];
#pragma unroll
      for (int i = 0; i < 8; i++) {
        afrag[mi][i]     = ap[hi * 16 + i];
        afrag[mi][8 + i] = ap[hi * 16 + 8 + i];
      }
    }
#pragma unroll
    for (int ni = 0; ni < 4; ni++) {
      bf16x16 bfrag;
      const __bf16* bp = &Ws[cur][(wn * 64 + ni * 16 + lane16) * 32 + hi * 16];
#pragma unroll
      for (int t = 0; t < 16; t++) bfrag[t] = bp[t];
#pragma unroll
      for (int mi = 0; mi < 2; mi++)
        acc[mi][ni] = __builtin_amdgcn_wmma_f32_16x16x32_bf16(
            false, afrag[mi], false, bfrag, (short)0, acc[mi][ni], false, false);
    }
    cur ^= 1;
  }

  for (int mi = 0; mi < 2; mi++)
    for (int ni = 0; ni < 4; ni++)
#pragma unroll
      for (int r = 0; r < 8; r++) {
        int rr = bm + wm * 32 + mi * 16 + hi * 8 + r;
        int cc = bn + wn * 64 + ni * 16 + lane16;
        C[(size_t)rr * N + cc] = acc[mi][ni][r];
      }
}

// ---------------- RoPE: fp32 [B,S,nh,128] -> bf16 [B,nh,S,128] ----------------
__global__ void rope_kernel(const float* __restrict__ in, __bf16* __restrict__ out, int nh) {
  int t = blockIdx.x * blockDim.x + threadIdx.x;
  int total = BATCH * SEQ * nh * 64;
  if (t >= total) return;
  int i = t & 63;
  int head = (t >> 6) % nh;
  int s = (t / (64 * nh)) % SEQ;
  int b = t / (64 * nh * SEQ);
  // inv_freq = theta^(-2i/128), theta = 10000
  float inv = __expf(-(float)(2 * i) * (9.210340371976184f / 128.0f));
  float fr = (float)s * inv;
  float sn, cs;
  __sincosf(fr, &sn, &cs);
  const float* p = in + ((size_t)(b * SEQ + s) * nh + head) * HD;
  float x1 = p[i], x2 = p[i + 64];
  __bf16* q = out + ((size_t)(b * nh + head) * SEQ + s) * HD;
  q[i]      = to_bf16(x1 * cs - x2 * sn);
  q[i + 64] = to_bf16(x2 * cs + x1 * sn);
}

// ---------------- V: fp32 [B,S,nkv,128] -> bf16 transposed [B,nkv,128,S] ----------------
__global__ void vtrans_kernel(const float* __restrict__ in, __bf16* __restrict__ out) {
  int t = blockIdx.x * blockDim.x + threadIdx.x;
  int total = BATCH * NKV * HD * SEQ;
  if (t >= total) return;
  int s = t % SEQ;
  int d = (t / SEQ) % HD;
  int kvh = (t / (SEQ * HD)) % NKV;
  int b = t / (SEQ * HD * NKV);
  out[((size_t)(b * NKV + kvh) * HD + d) * SEQ + s] =
      to_bf16(in[((size_t)(b * SEQ + s) * NKV + kvh) * HD + d]);
}

// ---------------- flash attention (causal, online softmax) ----------------
// grid = (S/128, B*NH); 8 waves x 16 query rows; K-tiles of 64 keys
__global__ __launch_bounds__(256)
void fattn_kernel(const __bf16* __restrict__ Q,   // [B,NH,S,HD]
                  const __bf16* __restrict__ Kc,  // [B,NKV,S,HD]
                  const __bf16* __restrict__ Vt,  // [B,NKV,HD,S]
                  __bf16* __restrict__ O) {       // [B,S,NH*HD]
  __shared__ __bf16 Ks[64 * 128];     // [key][d]
  __shared__ __bf16 Vs[128 * 64];     // [d][key]
  __shared__ __bf16 Ps[8][16][64];    // per-wave P tile (C-layout -> A-layout bounce)
  const int tid = threadIdx.x;
  const int lane = tid & 31, w = tid >> 5;
  const int lane16 = lane & 15, hi = lane >> 4;
  const int qb = blockIdx.x;
  const int bh = blockIdx.y;
  const int b = bh >> 5, h = bh & 31, kvh = h >> 2;

  // Q A-fragments, rows indexed by lane16
  const int qrow = qb * 128 + w * 16 + lane16;
  const __bf16* qp = Q + ((size_t)(b * NH + h) * SEQ + qrow) * HD;
  bf16x16 qf[4];
#pragma unroll
  for (int kk = 0; kk < 4; kk++)
#pragma unroll
    for (int i = 0; i < 8; i++) {
      qf[kk][i]     = qp[kk * 32 + hi * 16 + i];
      qf[kk][8 + i] = qp[kk * 32 + hi * 16 + 8 + i];
    }

  float mrow[8], lrow[8];
  f32x8 o[8];
  for (int r = 0; r < 8; r++) { mrow[r] = -1e30f; lrow[r] = 0.f; }
  for (int n = 0; n < 8; n++)
    for (int r = 0; r < 8; r++) o[n][r] = 0.f;

  const __bf16* kbase = Kc + (size_t)(b * NKV + kvh) * SEQ * HD;
  const __bf16* vbase = Vt + (size_t)(b * NKV + kvh) * HD * SEQ;

  const int nkt = 2 * qb + 2;
  for (int kt = 0; kt < nkt; ++kt) {
    __syncthreads();                    // previous iteration's consumers done
    // async DMA K tile (64x128) and V tile (128x64) into LDS
    for (int c = tid; c < 1024; c += 256) {
      int row = c >> 4, col = (c & 15) * 8;
      async_ld_b128(&Ks[row * 128 + col],
                    kbase + (size_t)(kt * 64 + row) * HD + col);
    }
    for (int c = tid; c < 1024; c += 256) {
      int row = c >> 3, col = (c & 7) * 8;
      async_ld_b128(&Vs[row * 64 + col],
                    vbase + (size_t)row * SEQ + kt * 64 + col);
    }
    wait_async0();
    __syncthreads();

    // S = Q K^T  (4 tiles of 16 keys, K-dim = 128 in 4 chunks)
    f32x8 sc[4];
#pragma unroll
    for (int j = 0; j < 4; j++) {
      f32x8 s;
      for (int e = 0; e < 8; e++) s[e] = 0.f;
#pragma unroll
      for (int kk = 0; kk < 4; kk++) {
        bf16x16 bf;
        const __bf16* bp = &Ks[(j * 16 + lane16) * 128 + kk * 32 + hi * 16];
#pragma unroll
        for (int t = 0; t < 16; t++) bf[t] = bp[t];
        s = __builtin_amdgcn_wmma_f32_16x16x32_bf16(false, qf[kk], false, bf,
                                                    (short)0, s, false, false);
      }
      sc[j] = s;
    }

    const float scale = 0.08838834764831845f;  // 1/sqrt(128)
    const bool domask = (kt >= 2 * qb);
    for (int j = 0; j < 4; j++)
      for (int r = 0; r < 8; r++) {
        float v = sc[j][r] * scale;
        if (domask) {
          int key = kt * 64 + j * 16 + lane16;
          int qg  = qb * 128 + w * 16 + hi * 8 + r;
          if (key > qg) v = -1e30f;
        }
        sc[j][r] = v;
      }

    // online softmax (rows live across lane16 within each 16-lane half)
    float corr[8];
    for (int r = 0; r < 8; r++) {
      float v = sc[0][r];
      for (int j = 1; j < 4; j++) v = fmaxf(v, sc[j][r]);
      for (int m = 1; m < 16; m <<= 1) v = fmaxf(v, __shfl_xor(v, m, 32));
      float mn = fmaxf(mrow[r], v);
      corr[r] = __expf(mrow[r] - mn);
      mrow[r] = mn;
    }
    for (int r = 0; r < 8; r++) {
      float sum = 0.f;
      for (int j = 0; j < 4; j++) {
        float p = __expf(sc[j][r] - mrow[r]);
        sc[j][r] = p;
        sum += p;
      }
      for (int m = 1; m < 16; m <<= 1) sum += __shfl_xor(sum, m, 32);
      lrow[r] = lrow[r] * corr[r] + sum;
    }
    for (int n = 0; n < 8; n++)
      for (int r = 0; r < 8; r++) o[n][r] *= corr[r];

    // P (C-layout) -> LDS -> A-layout
    for (int j = 0; j < 4; j++)
      for (int r = 0; r < 8; r++)
        Ps[w][hi * 8 + r][j * 16 + lane16] = to_bf16(sc[j][r]);

#pragma unroll
    for (int kk = 0; kk < 2; kk++) {
      bf16x16 pa;
      const __bf16* pp = &Ps[w][lane16][kk * 32];
#pragma unroll
      for (int i = 0; i < 8; i++) {
        pa[i]     = pp[hi * 16 + i];
        pa[8 + i] = pp[hi * 16 + 8 + i];
      }
#pragma unroll
      for (int n = 0; n < 8; n++) {
        bf16x16 bf;
        const __bf16* vp = &Vs[(n * 16 + lane16) * 64 + kk * 32 + hi * 16];
#pragma unroll
        for (int t = 0; t < 16; t++) bf[t] = vp[t];
        o[n] = __builtin_amdgcn_wmma_f32_16x16x32_bf16(false, pa, false, bf,
                                                       (short)0, o[n], false, false);
      }
    }
  }

  // normalize + write bf16 [B,S,HID]
  for (int r = 0; r < 8; r++) {
    float inv = 1.0f / lrow[r];
    int srow = qb * 128 + w * 16 + hi * 8 + r;
    __bf16* op = O + (size_t)(b * SEQ + srow) * HID + h * HD;
    for (int n = 0; n < 8; n++) op[n * 16 + lane16] = to_bf16(o[n][r] * inv);
  }
}

// ---------------- host-side orchestration ----------------
extern "C" void kernel_launch(void* const* d_in, const int* in_sizes, int n_in,
                              void* d_out, int out_size, void* d_ws, size_t ws_size,
                              hipStream_t stream) {
  const float* X  = (const float*)d_in[0];   // [B,S,H]
  const float* Wq = (const float*)d_in[1];   // [4096,4096]
  const float* Wk = (const float*)d_in[2];   // [1024,4096]
  const float* Wv = (const float*)d_in[3];   // [1024,4096]
  const float* Wo = (const float*)d_in[4];   // [4096,4096]
  float* out = (float*)d_out;

  char* ws = (char*)d_ws;
  size_t off = 0;
  auto alloc = [&](size_t bytes) -> char* {
    char* p = ws + off;
    off += (bytes + 255) & ~(size_t)255;
    return p;
  };
  const size_t M = MROWS;                 // 4096 rows of tokens
  __bf16* Xbf  = (__bf16*)alloc(M * HID * 2);
  __bf16* Wqb  = (__bf16*)alloc((size_t)HID * HID * 2);
  __bf16* Wkb  = (__bf16*)alloc((size_t)NKV * HD * HID * 2);
  __bf16* Wvb  = (__bf16*)alloc((size_t)NKV * HD * HID * 2);
  __bf16* Wob  = (__bf16*)alloc((size_t)HID * HID * 2);
  float*  Qf   = (float*)alloc(M * HID * 4);
  float*  Kf   = (float*)alloc(M * (size_t)(NKV * HD) * 4);
  float*  Vf   = (float*)alloc(M * (size_t)(NKV * HD) * 4);
  __bf16* Qbf  = (__bf16*)alloc(M * HID * 2);
  __bf16* Kbf  = (__bf16*)alloc(M * (size_t)(NKV * HD) * 2);
  __bf16* Vtb  = (__bf16*)alloc(M * (size_t)(NKV * HD) * 2);
  __bf16* Abf  = (__bf16*)alloc(M * HID * 2);

  const int TPB = 256;
  // 1) bf16 conversions
  {
    int n = (int)(M * HID);
    cvt_bf16_kernel<<<(n + TPB - 1) / TPB, TPB, 0, stream>>>(X, Xbf, n);
    n = HID * HID;
    cvt_bf16_kernel<<<(n + TPB - 1) / TPB, TPB, 0, stream>>>(Wq, Wqb, n);
    cvt_bf16_kernel<<<(n + TPB - 1) / TPB, TPB, 0, stream>>>(Wo, Wob, n);
    n = NKV * HD * HID;
    cvt_bf16_kernel<<<(n + TPB - 1) / TPB, TPB, 0, stream>>>(Wk, Wkb, n);
    cvt_bf16_kernel<<<(n + TPB - 1) / TPB, TPB, 0, stream>>>(Wv, Wvb, n);
  }
  // 2) projections
  gemm_bf16_kernel<<<dim3(HID / 128, M / 128), TPB, 0, stream>>>(Xbf, Wqb, Qf, (int)M, HID, HID);
  gemm_bf16_kernel<<<dim3(NKV * HD / 128, M / 128), TPB, 0, stream>>>(Xbf, Wkb, Kf, (int)M, NKV * HD, HID);
  gemm_bf16_kernel<<<dim3(NKV * HD / 128, M / 128), TPB, 0, stream>>>(Xbf, Wvb, Vf, (int)M, NKV * HD, HID);
  // 3) RoPE + V transpose
  {
    int n = BATCH * SEQ * NH * 64;
    rope_kernel<<<(n + TPB - 1) / TPB, TPB, 0, stream>>>(Qf, Qbf, NH);
    n = BATCH * SEQ * NKV * 64;
    rope_kernel<<<(n + TPB - 1) / TPB, TPB, 0, stream>>>(Kf, Kbf, NKV);
    n = BATCH * NKV * HD * SEQ;
    vtrans_kernel<<<(n + TPB - 1) / TPB, TPB, 0, stream>>>(Vf, Vtb);
  }
  // 4) flash attention
  fattn_kernel<<<dim3(SEQ / 128, BATCH * NH), TPB, 0, stream>>>(Qbf, Kbf, Vtb, Abf);
  // 5) output projection -> fp32 out
  gemm_bf16_kernel<<<dim3(HID / 128, M / 128), TPB, 0, stream>>>(Abf, Wob, out, (int)M, HID, HID);
}
